// AttentionLayer_20203526160658
// MI455X (gfx1250) — compile-verified, matrix operand order checked by reading, not run on hardware
//
#include <hip/hip_runtime.h>
#include <hip/hip_bf16.h>

typedef __attribute__((ext_vector_type(16))) _Float16 v16h;
typedef __attribute__((ext_vector_type(8)))  _Float16 v8h;
typedef __attribute__((ext_vector_type(8)))  float    v8f;

#define BATCH 8
#define SEQ   2048
#define DIM   512
#define DK    64
#define BS    (BATCH * SEQ)   // 16384

__device__ __forceinline__ v8f wmma16(v16h a, v16h b, v8f c) {
    return __builtin_amdgcn_wmma_f32_16x16x32_f16(
        /*neg_a=*/false, a, /*neg_b=*/false, b,
        /*c_mod=*/(short)0, c, /*reuse_a=*/false, /*reuse_b=*/false);
}

// Load a 16-bit WMMA operand fragment for one lane:
// elements 0..7  = mem[p .. p+7]      (K = h*8 .. h*8+7)
// elements 8..15 = mem[p+16 .. p+23]  (K = 16+h*8 .. 16+h*8+7)
// p must already include the h*8 lane offset.
__device__ __forceinline__ v16h ld16h(const _Float16* p) {
    v8h lo = *(const v8h*)(p);
    v8h hi = *(const v8h*)(p + 16);
    return __builtin_shufflevector(lo, hi, 0,1,2,3,4,5,6,7,8,9,10,11,12,13,14,15);
}

// Same fragment but converting from f32 memory on the fly.
__device__ __forceinline__ v16h ld16f(const float* p) {
    v16h a;
#pragma unroll
    for (int i = 0; i < 8; ++i) a[i]     = (_Float16)p[i];
#pragma unroll
    for (int i = 0; i < 8; ++i) a[8 + i] = (_Float16)p[16 + i];
    return a;
}

// ---------------------------------------------------------------------------
// Kernel 1: transpose + convert W_q/W_k/W_v [D,DK] f32 -> WT [3][DK][D] f16
// ---------------------------------------------------------------------------
__global__ void prep_wt(const float* __restrict__ Wq,
                        const float* __restrict__ Wk,
                        const float* __restrict__ Wv,
                        _Float16* __restrict__ WT) {
    int tid = blockIdx.x * blockDim.x + threadIdx.x;
    if (tid >= 3 * DK * DIM) return;
    int m   = tid / (DK * DIM);
    int rem = tid - m * (DK * DIM);
    int n   = rem >> 9;      // / DIM
    int d   = rem & (DIM - 1);
    const float* W = (m == 0) ? Wq : ((m == 1) ? Wk : Wv);
    WT[tid] = (_Float16)W[d * DK + n];
}

// ---------------------------------------------------------------------------
// Kernel 2: QKV projection.
//   Q  [BS,64] f16  (scale 1/sqrt(64) folded in)
//   K  [BS,64] f16
//   VT [B][64][S] f16  (V transposed so PV B-operand is K-contiguous)
// One wave = one (matrix, 16-row tile). 3*1024 waves, 256-thread blocks.
// ---------------------------------------------------------------------------
__global__ __launch_bounds__(256) void qkv_proj(const float* __restrict__ X,
                                                const _Float16* __restrict__ WT,
                                                _Float16* __restrict__ Qf,
                                                _Float16* __restrict__ Kf,
                                                _Float16* __restrict__ VT) {
    const int tid  = threadIdx.x;
    const int w    = tid >> 5;
    const int lane = tid & 31;
    const int gw   = blockIdx.x * 8 + w;       // 0..3071
    const int m    = gw >> 10;                 // 0=Q 1=K 2=V
    const int r0   = (gw & 1023) << 4;         // row tile base in [0,16384)
    const int n    = lane & 15;
    const int h    = lane >> 4;

    const float*    xrow = X  + (size_t)(r0 + n) * DIM;     // A row for this lane
    const _Float16* wtm  = WT + (size_t)m * DK * DIM;

    v8f acc[4] = {};
#pragma unroll 4
    for (int kc = 0; kc < 16; ++kc) {
        const int k0 = kc * 32;
        v16h a = ld16f(xrow + k0 + h * 8);
#pragma unroll
        for (int nt = 0; nt < 4; ++nt) {
            v16h b = ld16h(wtm + (size_t)(nt * 16 + n) * DIM + k0 + h * 8);
            acc[nt] = wmma16(a, b, acc[nt]);
        }
    }

#pragma unroll
    for (int nt = 0; nt < 4; ++nt) {
#pragma unroll
        for (int i = 0; i < 8; ++i) {
            const int M   = i + 8 * h;
            const int row = r0 + M;
            const int col = nt * 16 + n;
            const float v = acc[nt][i];
            if (m == 0) {
                Qf[(size_t)row * DK + col] = (_Float16)(v * 0.125f); // 1/sqrt(64)
            } else if (m == 1) {
                Kf[(size_t)row * DK + col] = (_Float16)v;
            } else {
                const int bb = row >> 11;          // / SEQ
                const int s  = row & (SEQ - 1);
                VT[((size_t)bb * DK + col) * SEQ + s] = (_Float16)v;
            }
        }
    }
}

// ---------------------------------------------------------------------------
// Kernel 3: fused attention. One block = (batch b, 16 queries).
//   scores (16x2048 f32) -> LDS; softmax; weights -> HBM once (f32);
//   probs compacted in-place to f16; PV via WMMA.
// LDS: 128KB scores + 4KB partial = 132KB (< 320KB/WGP).
// ---------------------------------------------------------------------------
__global__ __launch_bounds__(256) void attn_fused(const _Float16* __restrict__ Qf,
                                                  const _Float16* __restrict__ Kf,
                                                  const _Float16* __restrict__ VT,
                                                  float* __restrict__ outA,
                                                  float* __restrict__ outW) {
    __shared__ float sS[16][SEQ];     // 128 KB: scores, later f16 probs in low half of each row
    __shared__ float sAcc[16 * DK];   // 4 KB: PV partial from K-half 1

    const int b    = blockIdx.x >> 7;          // / 128
    const int q0   = (blockIdx.x & 127) << 4;
    const int tid  = threadIdx.x;
    const int w    = tid >> 5;                 // wave 0..7
    const int lane = tid & 31;
    const int n    = lane & 15;
    const int h    = lane >> 4;

    const _Float16* Qb = Qf + (size_t)(b * SEQ + q0) * DK;
    const _Float16* Kb = Kf + (size_t)b * SEQ * DK;
    const _Float16* Vb = VT + (size_t)b * DK * SEQ;

    // ---- Phase 1: scores tile row (16 x 2048) ----
    const _Float16* qrow = Qb + (size_t)n * DK;
    v16h aq0 = ld16h(qrow + 0  + h * 8);
    v16h aq1 = ld16h(qrow + 32 + h * 8);

#pragma unroll 2
    for (int j = 0; j < 16; ++j) {
        const int col0 = w * 256 + j * 16;
        const _Float16* krow = Kb + (size_t)(col0 + n) * DK;
        v16h bk0 = ld16h(krow + 0  + h * 8);
        v16h bk1 = ld16h(krow + 32 + h * 8);
        v8f c = {};
        c = wmma16(aq0, bk0, c);
        c = wmma16(aq1, bk1, c);
#pragma unroll
        for (int i = 0; i < 8; ++i) sS[i + 8 * h][col0 + n] = c[i];
    }
    __syncthreads();

    // ---- Phase 2: row softmax (wave handles rows 2w, 2w+1) ----
    // Values staged in registers so in-place f32 -> f16 compaction is race-free
    // (the whole wave finishes its reads before any lane writes).
#pragma unroll
    for (int rr = 0; rr < 2; ++rr) {
        const int r = w * 2 + rr;
        float vals[64];
        float mx = -1e30f;
#pragma unroll
        for (int i = 0; i < 64; ++i) {
            vals[i] = sS[r][lane + 32 * i];
            mx = fmaxf(mx, vals[i]);
        }
#pragma unroll
        for (int off = 16; off > 0; off >>= 1) mx = fmaxf(mx, __shfl_xor(mx, off, 32));
        float sum = 0.f;
#pragma unroll
        for (int i = 0; i < 64; ++i) {
            vals[i] = __expf(vals[i] - mx);
            sum += vals[i];
        }
#pragma unroll
        for (int off = 16; off > 0; off >>= 1) sum += __shfl_xor(sum, off, 32);
        const float inv = 1.0f / sum;

        _Float16* sPr = (_Float16*)&sS[r][0];
        float* wrow = outW + (size_t)(b * SEQ + q0 + r) * SEQ;
#pragma unroll
        for (int i = 0; i < 64; ++i) {
            const float p = vals[i] * inv;
            const int c = lane + 32 * i;
            sPr[c]  = (_Float16)p;   // f16 probs, low half of the row
            wrow[c] = p;             // weights output (f32, written once)
        }
    }
    __syncthreads();

    // ---- Phase 3: attended = P @ V. wave -> (dk-tile nt, K-half kh) ----
    const int nt = w & 3;
    const int kh = w >> 2;
    const _Float16* vrow = Vb + (size_t)(nt * 16 + n) * SEQ;
    const _Float16* sProw = (const _Float16*)&sS[n][0];

    v8f acc = {};
#pragma unroll 4
    for (int kc = kh * 32; kc < kh * 32 + 32; ++kc) {
        const int k0 = kc * 32;
        v16h ap = ld16h(sProw + k0 + h * 8);   // LDS f16 probs
        v16h bv = ld16h(vrow  + k0 + h * 8);   // V^T, K-contiguous
        acc = wmma16(ap, bv, acc);
    }

    if (kh == 1) {
#pragma unroll
        for (int i = 0; i < 8; ++i) sAcc[(i + 8 * h) * DK + nt * 16 + n] = acc[i];
    }
    __syncthreads();
    if (kh == 0) {
        float* orow = outA + (size_t)(b * SEQ + q0) * DK;
#pragma unroll
        for (int i = 0; i < 8; ++i) {
            const int M = i + 8 * h;
            orow[(size_t)M * DK + nt * 16 + n] = acc[i] + sAcc[M * DK + nt * 16 + n];
        }
    }
}

// ---------------------------------------------------------------------------
extern "C" void kernel_launch(void* const* d_in, const int* in_sizes, int n_in,
                              void* d_out, int out_size, void* d_ws, size_t ws_size,
                              hipStream_t stream) {
    (void)in_sizes; (void)n_in; (void)out_size; (void)ws_size;
    const float* X  = (const float*)d_in[0];
    const float* Wq = (const float*)d_in[1];
    const float* Wk = (const float*)d_in[2];
    const float* Wv = (const float*)d_in[3];

    char* ws = (char*)d_ws;
    _Float16* WT = (_Float16*)(ws);               // 3*64*512*2  = 0x30000 bytes
    _Float16* Qf = (_Float16*)(ws + 0x030000);    // 16384*64*2  = 0x200000
    _Float16* Kf = (_Float16*)(ws + 0x230000);    // 0x200000
    _Float16* VT = (_Float16*)(ws + 0x430000);    // 0x200000 (end 0x630000)

    float* outA = (float*)d_out;                              // [B,S,DK]
    float* outW = (float*)d_out + (size_t)BATCH * SEQ * DK;   // [B,S,S]

    prep_wt  <<<(3 * DK * DIM + 255) / 256, 256, 0, stream>>>(Wq, Wk, Wv, WT);
    qkv_proj <<<384, 256, 0, stream>>>(X, WT, Qf, Kf, VT);
    attn_fused<<<BATCH * (SEQ / 16), 256, 0, stream>>>(Qf, Kf, VT, outA, outW);
}
